// Encoder_3822520894033
// MI455X (gfx1250) — matrix operand order repaired and checked by reading, not compiled, not run on hardware
//
#include <hip/hip_runtime.h>
#include <hip/hip_bf16.h>

#define B_  256
#define T_  1024
#define F_  64
#define U_  128
#define NT_ 24          // 384 / 16 output column tiles

typedef __attribute__((ext_vector_type(16))) __bf16 v16bf;
typedef __attribute__((ext_vector_type(8)))  float  v8f;
typedef __attribute__((ext_vector_type(4)))  unsigned int u32x4;
typedef __attribute__((ext_vector_type(8)))  int          i32x8;
typedef __attribute__((ext_vector_type(4)))  int          i32x4;

static __device__ __forceinline__ v8f wmma_bf16(v16bf a, v16bf b, v8f c) {
  // D = A(16x32 bf16) * B(32x16 bf16) + C(16x16 f32)
  return __builtin_amdgcn_wmma_f32_16x16x32_bf16(false, a, false, b, (short)0, c,
                                                 false, false);
}

static __device__ __forceinline__ float fast_sigmoid(float x) {
  // v_exp_f32 + v_rcp_f32 (avoid IEEE division sequence on the scan critical path)
  return __builtin_amdgcn_rcpf(1.0f + __expf(-x));
}

// Gather an A-fragment (16x32 bf16, contiguous k = 16*hi + s) from an LDS tile stored
// row-major with padded leading dim `ldk` (pad => conflict-free banks).
static __device__ __forceinline__ v16bf load_a_frag(const __bf16* src, int ldk, int kt) {
  const int lane = threadIdx.x & 31;
  const int m  = lane & 15;
  const int hi = lane >> 4;
  v16bf a;
#pragma unroll
  for (int v = 0; v < 8; ++v) {
    // VGPR v holds K = k0, k0+1 (pairs); halves of the wave cover K offsets +16.
    const int k0 = kt * 32 + 2 * v + 16 * hi;
    const unsigned p = *(const unsigned*)(src + m * ldk + k0);  // two packed bf16
    ((unsigned*)&a)[v] = p;
  }
  return a;
}

// One-time load of a B-fragment (32x16) straight from f32 global weights
// (row-major [K,384]) into registers, converting to bf16.
static __device__ __forceinline__ v16bf load_b_frag_global(const float* Wsrc, int kt, int nt) {
  const int lane = threadIdx.x & 31;
  const int kbase = kt * 32 + (lane >> 4) * 16;
  const int n = nt * 16 + (lane & 15);
  v16bf r;
#pragma unroll
  for (int e = 0; e < 16; ++e) r[e] = (__bf16)Wsrc[(kbase + e) * 384 + n];
  return r;
}

// Per-step load of a pre-packed B-fragment: one contiguous 32B load per lane.
// (Loop-invariant args => compiler hoists these into registers for the scan.)
static __device__ __forceinline__ v16bf load_b_frag_packed(const __bf16* Wp, int kt, int nt) {
  const int lane = threadIdx.x & 31;
  return *(const v16bf*)(Wp + (((size_t)(kt * NT_ + nt)) * 32 + lane) * 16);
}

// Issue a TDM 2D-tile load: 16 rows x 128 bf16 from global (row stride T*U elems)
// into LDS at lds_byte_off, padding +1 DWORD after every 256B row so the LDS
// layout has a 130-element leading dim (bank-conflict-free for A gathers).
static __device__ __forceinline__ void tdm_load_tile(unsigned long long gaddr,
                                                     unsigned lds_byte_off) {
  u32x4 g0;
  g0[0] = 1u;                                   // count=1, user mode, no gather
  g0[1] = lds_byte_off;                         // lds_addr
  g0[2] = (unsigned)(gaddr & 0xFFFFFFFFull);    // global_addr[31:0]
  g0[3] = (unsigned)((gaddr >> 32) & 0x1FFFFFFull) | (2u << 30);  // addr[56:32]|type=2

  i32x8 g1;
  g1[0] = (1 << 16)        // data_size = 2 bytes
        | (1 << 20)        // pad_enable
        | (5 << 22);       // pad_interval: 2^(5+1)=64 DWORDs = 256B (one row)
                           // pad_amount bits 31:25 = 0 -> 1 DWORD pad
  g1[1] = (U_ & 0xFFFF) << 16;                  // tensor_dim0[15:0]=128
  g1[2] = (16 & 0xFFFF) << 16;                  // tensor_dim0 hi=0 | tensor_dim1=16
  g1[3] = (U_ & 0xFFFF) << 16;                  // tensor_dim1 hi=0 | tile_dim0=128
  g1[4] = 16;                                   // tile_dim1=16, tile_dim2=0
  g1[5] = T_ * U_;                              // tensor_dim0_stride (elements)
  g1[6] = 0;                                    // stride0 hi | stride1 lo
  g1[7] = 0;                                    // stride1 hi

  i32x4 g2 = {0, 0, 0, 0};                      // dims 2+ unused (tile_dim2=0)
  i32x4 g3 = {0, 0, 0, 0};
  i32x8 g4 = {0, 0, 0, 0, 0, 0, 0, 0};          // extra group (clang-23 6-arg form)

  __builtin_amdgcn_tensor_load_to_lds(g0, g1, g2, g3, g4, 0);
}

// Pack f32 weights [K,384] into bf16 WMMA B-fragment-major layout in workspace.
__global__ void pack_w_kernel(const float* __restrict__ W, __bf16* __restrict__ Wp, int KT) {
  const int total = KT * NT_ * 512;
  for (int idx = blockIdx.x * blockDim.x + threadIdx.x; idx < total;
       idx += gridDim.x * blockDim.x) {
    const int e    = idx & 15;
    const int lane = (idx >> 4) & 31;
    const int tile = idx >> 9;
    const int kt = tile / NT_, nt = tile % NT_;
    const int k = kt * 32 + (lane >> 4) * 16 + e;
    const int n = nt * 16 + (lane & 15);
    Wp[idx] = (__bf16)W[k * 384 + n];
  }
}

// One GRU layer scan. Each block owns 16 batch rows for all T steps.
// Wave wv owns hidden columns [16*wv, 16*wv+16): z cols n, r cols 128+n, h cols 256+n.
// KTX = number of 32-wide K tiles of the input (2 for F=64, 4 for U=128).
template <int KTX, bool XF32>
__global__ __launch_bounds__(256) void gru_scan_kernel(
    const void* __restrict__ xin,       // [B,T,KTX*32] f32 (layer1) or bf16 (layer2)
    const __bf16* __restrict__ Wpack,   // packed input-projection fragments
    const float* __restrict__ Uk,       // [U,384] recurrent weights f32
    const float* __restrict__ bias,     // [2,384]
    __bf16* __restrict__ seq_out,       // [B,T,U] bf16 or nullptr
    float* __restrict__ out_a,          // final state dest
    float* __restrict__ out_b) {        // optional duplicate dest
  constexpr int FIN  = KTX * 32;
  constexpr int LDX  = FIN + 2;         // padded LDS strides (bank-conflict-free)
  constexpr int LDH  = U_ + 2;
  constexpr int NBUF = XF32 ? 1 : 2;    // double-buffer x when TDM-fed (layer 2)

  __shared__ __bf16 sH[16 * LDH];
  __shared__ __bf16 sX[NBUF * 16 * LDX];

  const int tid  = threadIdx.x;
  const int lane = tid & 31;
  const int wv   = tid >> 5;            // wave 0..7 -> column group
  const int c    = lane & 15;           // output column within tile
  const int hi   = lane >> 4;
  const int r0   = blockIdx.x * 16;     // batch rows [r0, r0+16)

  // Per-lane biases (column-dependent only in C/D layout).
  const int nz = 16 * wv + c;
  const float bz  = bias[nz] + bias[384 + nz];
  const float br  = bias[128 + nz] + bias[384 + 128 + nz];
  const float bih = bias[256 + nz];
  const float brh = bias[384 + 256 + nz];

  // Recurrent weight fragments resident in VGPRs for the whole scan.
  v16bf Uz[4], Ur[4], Uh[4];
#pragma unroll
  for (int kt = 0; kt < 4; ++kt) {
    Uz[kt] = load_b_frag_global(Uk, kt, wv);
    Ur[kt] = load_b_frag_global(Uk, kt, 8 + wv);
    Uh[kt] = load_b_frag_global(Uk, kt, 16 + wv);
  }

  // h0 = 0
  for (int i = tid; i < 16 * LDH; i += 256) sH[i] = (__bf16)0.0f;
  v8f hold = {0.f, 0.f, 0.f, 0.f, 0.f, 0.f, 0.f, 0.f};

  // TDM pipeline prologue: kick off the t=0 activation tile (layer 2 only).
  if (!XF32) {
    if (wv == 0 && lane == 0)
      tdm_load_tile((unsigned long long)(uintptr_t)xin +
                        (unsigned long long)r0 * T_ * U_ * 2,
                    (unsigned)(size_t)(void*)&sX[0]);
  }
  __syncthreads();

  for (int t = 0; t < T_; ++t) {
    const __bf16* sXcur = &sX[(XF32 ? 0 : (t & 1)) * 16 * LDX];

    // ---- stage x_t into LDS (bf16) ----
    if (XF32) {
      const float* xf = (const float*)xin;
      for (int i = tid; i < 16 * FIN; i += 256) {
        const int rr = i / FIN, k = i - rr * FIN;
        sX[rr * LDX + k] = (__bf16)xf[((size_t)(r0 + rr) * T_ + t) * FIN + k];
      }
      if (t + 1 < T_)
        __builtin_prefetch(xf + ((size_t)(r0 + (tid & 15)) * T_ + (t + 1)) * FIN, 0, 1);
      __syncthreads();  // x_t staged AND previous h write visible
    } else {
      // Wait for the TDM tile covering step t, publish it, then immediately
      // issue the TDM for step t+1 into the other buffer (overlaps the WMMAs).
      if (wv == 0) __builtin_amdgcn_s_wait_tensorcnt(0);
      __syncthreads();  // TDM data + previous h write visible to all waves
      if (wv == 0 && lane == 0 && (t + 1 < T_))
        tdm_load_tile((unsigned long long)(uintptr_t)xin +
                          ((unsigned long long)r0 * T_ + (t + 1)) * U_ * 2,
                      (unsigned)(size_t)(void*)&sX[((t + 1) & 1) * 16 * LDX]);
    }

    // ---- matrix work: acc = x@W (+ h@U merged for z,r; split for h-gate) ----
    v8f az  = {0.f, 0.f, 0.f, 0.f, 0.f, 0.f, 0.f, 0.f};
    v8f ar  = az, axh = az, arh = az;
#pragma unroll
    for (int kt = 0; kt < KTX; ++kt) {
      const v16bf ax = load_a_frag(sXcur, LDX, kt);
      az  = wmma_bf16(ax, load_b_frag_packed(Wpack, kt, wv),      az);
      ar  = wmma_bf16(ax, load_b_frag_packed(Wpack, kt, 8 + wv),  ar);
      axh = wmma_bf16(ax, load_b_frag_packed(Wpack, kt, 16 + wv), axh);
    }
#pragma unroll
    for (int kt = 0; kt < 4; ++kt) {
      const v16bf ah = load_a_frag(sH, LDH, kt);
      az  = wmma_bf16(ah, Uz[kt], az);
      ar  = wmma_bf16(ah, Ur[kt], ar);
      arh = wmma_bf16(ah, Uh[kt], arh);
    }

    // ---- gates (C/D layout: vgpr i -> row i+8*hi, col c) ----
    v8f hnew;
#pragma unroll
    for (int i = 0; i < 8; ++i) {
      const float z  = fast_sigmoid(az[i] + bz);
      const float rg = fast_sigmoid(ar[i] + br);
      const float hh = fmaxf(0.0f, (axh[i] + bih) + rg * (arh[i] + brh));
      hnew[i] = z * hold[i] + (1.0f - z) * hh;
    }
    hold = hnew;

    __syncthreads();  // all reads of sH complete before overwrite
#pragma unroll
    for (int i = 0; i < 8; ++i) {
      const int row = i + 8 * hi;
      const __bf16 hv = (__bf16)hnew[i];
      sH[row * LDH + 16 * wv + c] = hv;
      if (seq_out)
        seq_out[((size_t)(r0 + row) * T_ + t) * U_ + 16 * wv + c] = hv;
    }
  }

  // final state(s), f32
#pragma unroll
  for (int i = 0; i < 8; ++i) {
    const int row = i + 8 * hi;
    const float v = hold[i];
    out_a[(r0 + row) * U_ + 16 * wv + c] = v;
    if (out_b) out_b[(r0 + row) * U_ + 16 * wv + c] = v;
  }
}

extern "C" void kernel_launch(void* const* d_in, const int* in_sizes, int n_in,
                              void* d_out, int out_size, void* d_ws, size_t ws_size,
                              hipStream_t stream) {
  (void)in_sizes; (void)n_in; (void)out_size; (void)ws_size;
  const float* x  = (const float*)d_in[0];  // [B,T,F]
  const float* W1 = (const float*)d_in[1];  // [F,384]
  const float* U1 = (const float*)d_in[2];  // [U,384]
  const float* b1 = (const float*)d_in[3];  // [2,384]
  const float* W2 = (const float*)d_in[4];  // [U,384]
  const float* U2 = (const float*)d_in[5];  // [U,384]
  const float* b2 = (const float*)d_in[6];  // [2,384]
  float* out = (float*)d_out;               // [x | state1 | state2], each B*U

  char* ws = (char*)d_ws;
  __bf16* seq1 = (__bf16*)ws;                                   // 64 MB
  __bf16* W1p  = (__bf16*)(ws + (size_t)B_ * T_ * U_ * 2);      // 48 KB
  __bf16* W2p  = W1p + (size_t)2 * NT_ * 512;                   // 96 KB

  pack_w_kernel<<<96, 256, 0, stream>>>(W1, W1p, 2);
  pack_w_kernel<<<192, 256, 0, stream>>>(W2, W2p, 4);

  // Layer 1: reads f32 input, writes seq1 (bf16) + state1.
  gru_scan_kernel<2, true><<<B_ / 16, 256, 0, stream>>>(
      (const void*)x, W1p, U1, b1, seq1, out + B_ * U_, nullptr);

  // Layer 2: reads seq1 via TDM double-buffer, writes x (out[0]) and state2.
  gru_scan_kernel<4, false><<<B_ / 16, 256, 0, stream>>>(
      (const void*)seq1, W2p, U2, b2, nullptr, out, out + 2 * B_ * U_);
}
